// TRAG_9002251453233
// MI455X (gfx1250) — matrix operand order repaired and checked by reading, not compiled, not run on hardware
//
#include <hip/hip_runtime.h>
#include <hip/hip_bf16.h>
#include <math.h>

// Problem constants
#define BB    16
#define TT    8
#define CC    2048
#define C8N   256
#define C4N   512
#define HWN   128
#define NPN   4
#define NIMG  (BB*TT)    // 128
#define NPAIR (BB*NPN)   // 64

typedef __attribute__((ext_vector_type(16))) __bf16 bf16x16;
typedef __attribute__((ext_vector_type(8)))  float  f32x8;

__device__ __forceinline__ __bf16 bfraw(unsigned short s) {
  union { unsigned short s; __bf16 h; } cv; cv.s = s; return cv.h;
}
// Native conversions: let the backend pick v_cvt-class bf16 instructions.
__device__ __forceinline__ __bf16 mkbf(float f) { return (__bf16)f; }
__device__ __forceinline__ unsigned pk2(float lo, float hi) {
  union { struct { __bf16 a, b; } s; unsigned u; } cv;
  cv.s.a = (__bf16)lo; cv.s.b = (__bf16)hi;
  return cv.u;
}
__device__ __forceinline__ f32x8 wmma_bf16(bf16x16 a, bf16x16 b, f32x8 c) {
  return __builtin_amdgcn_wmma_f32_16x16x32_bf16(false, a, false, b, (short)0, c, false, false);
}
__device__ __forceinline__ float sigm(float z) { return 1.0f / (1.0f + __expf(-z)); }

// Load a 16x32 bf16 A-operand tile row (contiguous k = 16*half + s) as two b128s.
__device__ __forceinline__ bf16x16 load_a_bf16(const __bf16* __restrict__ rowptr,
                                               int kb, int half) {
  union { uint4 q[2]; bf16x16 v; } u;
  u.q[0] = *(const uint4*)(rowptr + kb + half * 16);
  u.q[1] = *(const uint4*)(rowptr + kb + half * 16 + 8);
  return u.v;
}

// ---------------------------------------------------------------------------
// cvt: fp32 -> bf16, 4 elements per thread
// ---------------------------------------------------------------------------
__global__ __launch_bounds__(256) void k_cvt(const float* __restrict__ src,
                                             __bf16* __restrict__ dst, int n4) {
  int i = blockIdx.x * blockDim.x + threadIdx.x;
  if (i >= n4) return;
  float4 v = ((const float4*)src)[i];
  uint2 r; r.x = pk2(v.x, v.y); r.y = pk2(v.z, v.w);
  ((uint2*)dst)[i] = r;
}

// ---------------------------------------------------------------------------
// K0a: cp = relu(theta_w @ vect + b) split into c0[row,o], c1[row,o]
// ---------------------------------------------------------------------------
__global__ __launch_bounds__(256) void k_cp(const float* __restrict__ theta_w,
                                            const float* __restrict__ theta_b,
                                            const float* __restrict__ vect,
                                            float* __restrict__ c0, float* __restrict__ c1) {
  int idx = blockIdx.x * blockDim.x + threadIdx.x;          // 32768
  if (idx >= BB * C8N * TT) return;
  int t = idx & 7, o = (idx >> 3) & 255, b = idx >> 11;
  const float* v = vect + (size_t)(b * TT + t) * CC;
  const float* w = theta_w + (size_t)o * CC;
  float acc = theta_b[o];
  for (int c = 0; c < CC; c += 4) {
    float4 vv = *(const float4*)(v + c);
    float4 ww = *(const float4*)(w + c);
    acc += vv.x * ww.x + vv.y * ww.y + vv.z * ww.z + vv.w * ww.w;
  }
  acc = fmaxf(acc, 0.0f);
  int row = b * NPN + (t >> 1);
  if (t & 1) c1[row * C8N + o] = acc; else c0[row * C8N + o] = acc;
}

// ---------------------------------------------------------------------------
// K0b: h00 = relu([c0,c1] @ w1^T + b1), h01 = relu([c1,c0] @ w1^T + b1)
// ---------------------------------------------------------------------------
__global__ __launch_bounds__(256) void k_mlp1(const float* __restrict__ w1,
                                              const float* __restrict__ b1,
                                              const float* __restrict__ c0,
                                              const float* __restrict__ c1,
                                              float* __restrict__ h00, float* __restrict__ h01) {
  int idx = blockIdx.x * blockDim.x + threadIdx.x;          // 64*256
  if (idx >= NPAIR * C8N) return;
  int j = idx & 255, row = idx >> 8;
  const float* w = w1 + (size_t)j * C4N;
  float a0 = b1[j], a1 = b1[j];
  for (int i = 0; i < C8N; ++i) {
    float wa = w[i], wb = w[C8N + i];
    float x0 = c0[row * C8N + i], x1 = c1[row * C8N + i];
    a0 += wa * x0 + wb * x1;
    a1 += wa * x1 + wb * x0;
  }
  h00[idx] = fmaxf(a0, 0.0f);
  h01[idx] = fmaxf(a1, 0.0f);
}

// ---------------------------------------------------------------------------
// K0c: p00 = sigmoid(h00 @ w2^T + b2), p01 likewise
// ---------------------------------------------------------------------------
__global__ __launch_bounds__(256) void k_mlp2(const float* __restrict__ w2,
                                              const float* __restrict__ b2,
                                              const float* __restrict__ h00,
                                              const float* __restrict__ h01,
                                              float* __restrict__ p00, float* __restrict__ p01) {
  int idx = blockIdx.x * blockDim.x + threadIdx.x;          // 64*2048
  if (idx >= NPAIR * CC) return;
  int o = idx & 2047, row = idx >> 11;
  const float* w = w2 + (size_t)o * C8N;
  float a0 = b2[o], a1 = b2[o];
  for (int i = 0; i < C8N; ++i) {
    float wv = w[i];
    a0 += wv * h00[row * C8N + i];
    a1 += wv * h01[row * C8N + i];
  }
  p00[idx] = sigm(a0);
  p01[idx] = sigm(a1);
}

// ---------------------------------------------------------------------------
// K2: fused gamma+beta CBR GEMM.  LDS-staged X (double-buffered, bf16, [hw][k]
// transposed so a lane's B operand is two ds_load_b128).  Each wave computes
// TWO 16-row M-tiles for both gamma and beta -> 4 WMMAs per staged B operand.
// ---------------------------------------------------------------------------
__global__ __launch_bounds__(256) void k_gamma_beta(
    const float* __restrict__ x,
    const __bf16* __restrict__ gwb, const __bf16* __restrict__ bwb,
    const float* __restrict__ gs, const float* __restrict__ gb,
    const float* __restrict__ bs, const float* __restrict__ bb,
    __bf16* __restrict__ gammaT, __bf16* __restrict__ betaN) {
  __shared__ __align__(16) unsigned short xs[2][HWN * 32];   // 2 x 8KB, [hw][k]

  const int n  = blockIdx.x;            // image 0..127
  const int m0 = blockIdx.y * 32;       // 32 o-channels per block (2 m-tiles)
  const int tid  = threadIdx.x;
  const int lane = tid & 31;
  const int wave = tid >> 5;            // 8 waves -> hw tiles 0..7
  const int hw0  = wave * 16;
  const int lm   = lane & 15;
  const int half = lane >> 4;

  const float* xn = x + (size_t)n * CC * HWN;
  const __bf16* garow0 = gwb + (size_t)(m0 + lm) * CC;
  const __bf16* garow1 = gwb + (size_t)(m0 + 16 + lm) * CC;
  const __bf16* barow0 = bwb + (size_t)(m0 + lm) * CC;
  const __bf16* barow1 = bwb + (size_t)(m0 + 16 + lm) * CC;

  // staging decomposition: thread -> k-pair, 8-hw group
  const int kp  = tid & 15;             // k pair within 32-slab
  const int hwg = (tid >> 4) * 8;       // hw group of 8

  f32x8 accg0 = {}, accg1 = {}, accb0 = {}, accb1 = {};

  auto stage = [&](int kb, int buf) {
    const float* r0 = xn + (size_t)(kb + 2 * kp) * HWN + hwg;
    float4 x00 = *(const float4*)(r0);
    float4 x01 = *(const float4*)(r0 + 4);
    float4 x10 = *(const float4*)(r0 + HWN);
    float4 x11 = *(const float4*)(r0 + HWN + 4);
    __builtin_prefetch(r0 + 64 * HWN, 0, 0);
    unsigned* d = (unsigned*)&xs[buf][0];      // u32 index = hw*16 + kp
    d[(hwg + 0) * 16 + kp] = pk2(x00.x, x10.x);
    d[(hwg + 1) * 16 + kp] = pk2(x00.y, x10.y);
    d[(hwg + 2) * 16 + kp] = pk2(x00.z, x10.z);
    d[(hwg + 3) * 16 + kp] = pk2(x00.w, x10.w);
    d[(hwg + 4) * 16 + kp] = pk2(x01.x, x11.x);
    d[(hwg + 5) * 16 + kp] = pk2(x01.y, x11.y);
    d[(hwg + 6) * 16 + kp] = pk2(x01.z, x11.z);
    d[(hwg + 7) * 16 + kp] = pk2(x01.w, x11.w);
  };

  stage(0, 0);
  __syncthreads();

  const uint4* lq0 = (const uint4*)&xs[0][0];
  const uint4* lq1 = (const uint4*)&xs[1][0];
  const int bidx = (hw0 + lm) * 4 + half * 2;   // uint4 index of this lane's B

  for (int it = 0; it < CC / 32; ++it) {
    const int kb = it * 32;
    if (it + 1 < CC / 32) stage(kb + 32, (it + 1) & 1);

    bf16x16 ag0 = load_a_bf16(garow0, kb, half);
    bf16x16 ag1 = load_a_bf16(garow1, kb, half);
    bf16x16 ab0 = load_a_bf16(barow0, kb, half);
    bf16x16 ab1 = load_a_bf16(barow1, kb, half);

    const uint4* lq = (it & 1) ? lq1 : lq0;
    union { uint4 q[2]; bf16x16 v; } ub;
    ub.q[0] = lq[bidx];
    ub.q[1] = lq[bidx + 1];

    accg0 = wmma_bf16(ag0, ub.v, accg0);
    accb0 = wmma_bf16(ab0, ub.v, accb0);
    accg1 = wmma_bf16(ag1, ub.v, accg1);
    accb1 = wmma_bf16(ab1, ub.v, accb1);
    __syncthreads();
  }

  // Epilogue: relu(y*s+b); gamma stored transposed (hw,o), beta normal (o,hw)
  const int hw = hw0 + lm;
  #pragma unroll
  for (int r = 0; r < 8; ++r) {
    int ma = m0 + r + 8 * half;
    int mb = ma + 16;
    float yg0 = fmaxf(accg0[r] * gs[ma] + gb[ma], 0.0f);
    float yg1 = fmaxf(accg1[r] * gs[mb] + gb[mb], 0.0f);
    float yb0 = fmaxf(accb0[r] * bs[ma] + bb[ma], 0.0f);
    float yb1 = fmaxf(accb1[r] * bs[mb] + bb[mb], 0.0f);
    gammaT[((size_t)n * HWN + hw) * C8N + ma] = mkbf(yg0);
    gammaT[((size_t)n * HWN + hw) * C8N + mb] = mkbf(yg1);
    betaN [((size_t)n * C8N + ma) * HWN + hw] = mkbf(yb0);
    betaN [((size_t)n * C8N + mb) * HWN + hw] = mkbf(yb1);
  }
}

// ---------------------------------------------------------------------------
// K3: Gs0[s,t]=sum_c g_even[c,s]*b_odd[c,t]; Gs1 swapped.  Raw bf16 in/out,
// stored in both [s,t] and [t,s] orders for downstream operand roles.
// ---------------------------------------------------------------------------
__global__ __launch_bounds__(256) void k_gs(
    const __bf16* __restrict__ gammaT, const __bf16* __restrict__ betaN,
    __bf16* __restrict__ Gs0n, __bf16* __restrict__ Gs0t,
    __bf16* __restrict__ Gs1n, __bf16* __restrict__ Gs1t) {
  const int np = blockIdx.x;             // pair 0..63
  const int b = np >> 2, p = np & 3;
  const int ne = b * TT + 2 * p, no = ne + 1;
  const int lane = threadIdx.x & 31, wave = threadIdx.x >> 5;
  const int lm = lane & 15, half = lane >> 4;
  const int s0 = blockIdx.y * 16;        // m tile over s
  const int t0 = wave * 16;              // n tile over t
  const __bf16* gTe = gammaT + (size_t)ne * HWN * C8N + (size_t)(s0 + lm) * C8N;
  const __bf16* gTo = gammaT + (size_t)no * HWN * C8N + (size_t)(s0 + lm) * C8N;
  const __bf16* bNe = betaN  + (size_t)ne * C8N * HWN;
  const __bf16* bNo = betaN  + (size_t)no * C8N * HWN;
  f32x8 a0 = {}; f32x8 a1 = {};

  for (int kb = 0; kb < C8N; kb += 32) {
    bf16x16 A0 = load_a_bf16(gTe, kb, half);
    bf16x16 A1 = load_a_bf16(gTo, kb, half);
    bf16x16 B0, B1;
    #pragma unroll
    for (int e = 0; e < 16; ++e) {
      int k = kb + half * 16 + e;
      B0[e] = bNo[(size_t)k * HWN + t0 + lm];   // Gs0: gamma_even x beta_odd
      B1[e] = bNe[(size_t)k * HWN + t0 + lm];   // Gs1: gamma_odd  x beta_even
    }
    a0 = wmma_bf16(A0, B0, a0);
    a1 = wmma_bf16(A1, B1, a1);
  }

  __bf16* g0n = Gs0n + (size_t)np * HWN * HWN;
  __bf16* g0t = Gs0t + (size_t)np * HWN * HWN;
  __bf16* g1n = Gs1n + (size_t)np * HWN * HWN;
  __bf16* g1t = Gs1t + (size_t)np * HWN * HWN;
  #pragma unroll
  for (int r = 0; r < 8; ++r) {
    int s = s0 + r + 8 * half, t = t0 + lm;
    __bf16 v0 = mkbf(a0[r]), v1 = mkbf(a1[r]);
    g0n[s * HWN + t] = v0;  g0t[t * HWN + s] = v0;
    g1n[s * HWN + t] = v1;  g1t[t * HWN + s] = v1;
  }
}

// ---------------------------------------------------------------------------
// K4: gj0/gj1 = relu(gg CBR over j0/j1).  A = gg_w bf16 (shared), B piecewise
// from the stored Gs orders (lane-coalesced bf16 rows).
// ---------------------------------------------------------------------------
__global__ __launch_bounds__(256) void k_gg(
    const __bf16* __restrict__ ggwb, const float* __restrict__ ggs, const float* __restrict__ ggb,
    const __bf16* __restrict__ Gs0n, const __bf16* __restrict__ Gs0t,
    const __bf16* __restrict__ Gs1n, const __bf16* __restrict__ Gs1t,
    float* __restrict__ gj0, float* __restrict__ gj1) {
  const int np = blockIdx.x;
  const int m0 = blockIdx.y * 16;        // o tile (8 of 128)
  const int lane = threadIdx.x & 31, wave = threadIdx.x >> 5;
  const int lm = lane & 15, half = lane >> 4;
  const int sp0 = wave * 16;
  const size_t base = (size_t)np * HWN * HWN;
  const __bf16* s00 = Gs0t + base;  const __bf16* s01 = Gs1n + base;  // j0
  const __bf16* s10 = Gs1t + base;  const __bf16* s11 = Gs0n + base;  // j1
  const __bf16* arow = ggwb + (size_t)(m0 + lm) * 2 * HWN;
  f32x8 a0 = {}; f32x8 a1 = {};

  for (int kb = 0; kb < 2 * HWN; kb += 32) {
    bf16x16 A = load_a_bf16(arow, kb, half);
    const __bf16* p0 = (kb < HWN) ? s00 : s01;
    const __bf16* p1 = (kb < HWN) ? s10 : s11;
    bf16x16 B0, B1;
    #pragma unroll
    for (int e = 0; e < 16; ++e) {
      int kk = (kb + half * 16 + e) & (HWN - 1);
      B0[e] = p0[(size_t)kk * HWN + sp0 + lm];
      B1[e] = p1[(size_t)kk * HWN + sp0 + lm];
    }
    a0 = wmma_bf16(A, B0, a0);
    a1 = wmma_bf16(A, B1, a1);
  }
  #pragma unroll
  for (int r = 0; r < 8; ++r) {
    int m = m0 + r + 8 * half, sp = sp0 + lm;
    gj0[base + (size_t)m * HWN + sp] = fmaxf(a0[r] * ggs[m] + ggb[m], 0.0f);
    gj1[base + (size_t)m * HWN + sp] = fmaxf(a1[r] * ggs[m] + ggb[m], 0.0f);
  }
}

// ---------------------------------------------------------------------------
// K5: ta/tb = relu(tte CBR over [e0,e1]/[e1,e0]).  bf16 A and B (pre-conv).
// ---------------------------------------------------------------------------
__global__ __launch_bounds__(256) void k_tte(
    const __bf16* __restrict__ ttewb, const float* __restrict__ ts, const float* __restrict__ tb_,
    const __bf16* __restrict__ embedb, float* __restrict__ ta, float* __restrict__ tbo) {
  const int np = blockIdx.x;
  const int b = np >> 2, p = np & 3;
  const __bf16* e0 = embedb + (size_t)(b * TT + 2 * p)     * 128 * HWN;
  const __bf16* e1 = embedb + (size_t)(b * TT + 2 * p + 1) * 128 * HWN;
  const int m0 = blockIdx.y * 16;
  const int lane = threadIdx.x & 31, wave = threadIdx.x >> 5;
  const int lm = lane & 15, half = lane >> 4;
  const int sp0 = wave * 16;
  const __bf16* arow = ttewb + (size_t)(m0 + lm) * 256;
  f32x8 a0 = {}; f32x8 a1 = {};

  for (int kb = 0; kb < 256; kb += 32) {
    bf16x16 A = load_a_bf16(arow, kb, half);
    const __bf16* pA = (kb < 128) ? e0 : e1;   // ta: [e0,e1]
    const __bf16* pB = (kb < 128) ? e1 : e0;   // tb: [e1,e0]
    bf16x16 B0, B1;
    #pragma unroll
    for (int e = 0; e < 16; ++e) {
      int kk = (kb + half * 16 + e) & 127;
      B0[e] = pA[(size_t)kk * HWN + sp0 + lm];
      B1[e] = pB[(size_t)kk * HWN + sp0 + lm];
    }
    a0 = wmma_bf16(A, B0, a0);
    a1 = wmma_bf16(A, B1, a1);
  }
  const size_t base = (size_t)np * 128 * HWN;
  #pragma unroll
  for (int r = 0; r < 8; ++r) {
    int m = m0 + r + 8 * half, sp = sp0 + lm;
    ta [base + (size_t)m * HWN + sp] = fmaxf(a0[r] * ts[m] + tb_[m], 0.0f);
    tbo[base + (size_t)m * HWN + sp] = fmaxf(a1[r] * ts[m] + tb_[m], 0.0f);
  }
}

// ---------------------------------------------------------------------------
// K6: per-pixel gate: pa = sigmoid((te_w . [ta;gj0]) * s + b), pb likewise
// ---------------------------------------------------------------------------
__global__ __launch_bounds__(256) void k_gate(
    const float* __restrict__ tew, const float* __restrict__ tes, const float* __restrict__ teb,
    const float* __restrict__ ta, const float* __restrict__ tb,
    const float* __restrict__ gj0, const float* __restrict__ gj1,
    float* __restrict__ pa, float* __restrict__ pb) {
  int idx = blockIdx.x * blockDim.x + threadIdx.x;   // 64*128
  if (idx >= NPAIR * HWN) return;
  int np = idx >> 7, sp = idx & 127;
  const size_t base = (size_t)np * 128 * HWN + sp;
  float y0 = 0.0f, y1 = 0.0f;
  for (int c = 0; c < 128; ++c) {
    float wA = tew[c], wB = tew[128 + c];
    y0 += wA * ta[base + (size_t)c * HWN] + wB * gj0[base + (size_t)c * HWN];
    y1 += wA * tb[base + (size_t)c * HWN] + wB * gj1[base + (size_t)c * HWN];
  }
  float s = tes[0], bb = teb[0];
  pa[idx] = sigm(y0 * s + bb);
  pb[idx] = sigm(y1 * s + bb);
}

// ---------------------------------------------------------------------------
// K7: out = relu(p00*pa*f0 + p01*pb*f1)^2   (memory-bound, float4 vectorized)
// ---------------------------------------------------------------------------
__global__ __launch_bounds__(256) void k_out(
    const float* __restrict__ featmap,
    const float* __restrict__ p00, const float* __restrict__ p01,
    const float* __restrict__ pa, const float* __restrict__ pb,
    float* __restrict__ out) {
  int idx = blockIdx.x * blockDim.x + threadIdx.x;   // 4.19M float4s
  int hw = (idx & 31) * 4;
  int c  = (idx >> 5) & 2047;
  int np = idx >> 16;
  int b = np >> 2, p = np & 3;
  const float4 f0 = *(const float4*)(featmap + (((size_t)(b * TT + 2 * p)     * CC + c) * HWN + hw));
  const float4 f1 = *(const float4*)(featmap + (((size_t)(b * TT + 2 * p + 1) * CC + c) * HWN + hw));
  float g0 = p00[(size_t)np * CC + c];
  float g1 = p01[(size_t)np * CC + c];
  const float4 a4 = *(const float4*)(pa + (size_t)np * HWN + hw);
  const float4 b4 = *(const float4*)(pb + (size_t)np * HWN + hw);
  float4 r;
  r.x = fmaxf(g0 * a4.x * f0.x + g1 * b4.x * f1.x, 0.0f);
  r.y = fmaxf(g0 * a4.y * f0.y + g1 * b4.y * f1.y, 0.0f);
  r.z = fmaxf(g0 * a4.z * f0.z + g1 * b4.z * f1.z, 0.0f);
  r.w = fmaxf(g0 * a4.w * f0.w + g1 * b4.w * f1.w, 0.0f);
  r.x *= r.x; r.y *= r.y; r.z *= r.z; r.w *= r.w;
  *(float4*)(out + (size_t)idx * 4) = r;
}

// ---------------------------------------------------------------------------
extern "C" void kernel_launch(void* const* d_in, const int* in_sizes, int n_in,
                              void* d_out, int out_size, void* d_ws, size_t ws_size,
                              hipStream_t stream) {
  const float* featmap  = (const float*)d_in[0];
  const float* re_fm    = (const float*)d_in[1];
  const float* vect     = (const float*)d_in[2];
  const float* embed    = (const float*)d_in[3];
  const float* gamma_w  = (const float*)d_in[4];
  const float* gamma_s  = (const float*)d_in[5];
  const float* gamma_b  = (const float*)d_in[6];
  const float* beta_w   = (const float*)d_in[7];
  const float* beta_s   = (const float*)d_in[8];
  const float* beta_b   = (const float*)d_in[9];
  const float* gg_w     = (const float*)d_in[10];
  const float* gg_s     = (const float*)d_in[11];
  const float* gg_b     = (const float*)d_in[12];
  const float* tte_w    = (const float*)d_in[13];
  const float* tte_s    = (const float*)d_in[14];
  const float* tte_b    = (const float*)d_in[15];
  const float* te_w     = (const float*)d_in[16];
  const float* te_s     = (const float*)d_in[17];
  const float* te_b     = (const float*)d_in[18];
  const float* theta_w  = (const float*)d_in[19];
  const float* theta_b  = (const float*)d_in[20];
  const float* cp_w1    = (const float*)d_in[21];
  const float* cp_b1    = (const float*)d_in[22];
  const float* cp_w2    = (const float*)d_in[23];
  const float* cp_b2    = (const float*)d_in[24];
  float* out = (float*)d_out;

  // Workspace carve-up (256B-aligned)
  char* w = (char*)d_ws;
  size_t off = 0;
  auto carve = [&](size_t bytes) { char* p = w + off; off += (bytes + 255) & ~(size_t)255; return p; };
  float*  c0    = (float*)carve((size_t)NPAIR * C8N * 4);
  float*  c1    = (float*)carve((size_t)NPAIR * C8N * 4);
  float*  h00   = (float*)carve((size_t)NPAIR * C8N * 4);
  float*  h01   = (float*)carve((size_t)NPAIR * C8N * 4);
  float*  p00   = (float*)carve((size_t)NPAIR * CC * 4);
  float*  p01   = (float*)carve((size_t)NPAIR * CC * 4);
  __bf16* gwb   = (__bf16*)carve((size_t)C8N * CC * 2);           // gamma_w bf16
  __bf16* bwb   = (__bf16*)carve((size_t)C8N * CC * 2);           // beta_w bf16
  __bf16* ggwb  = (__bf16*)carve((size_t)128 * 256 * 2);          // gg_w bf16
  __bf16* ttewb = (__bf16*)carve((size_t)128 * 256 * 2);          // tte_w bf16
  __bf16* embedb= (__bf16*)carve((size_t)NIMG * 128 * HWN * 2);   // embed bf16
  __bf16* gammaT= (__bf16*)carve((size_t)NIMG * HWN * C8N * 2);
  __bf16* betaN = (__bf16*)carve((size_t)NIMG * C8N * HWN * 2);
  __bf16* Gs0n  = (__bf16*)carve((size_t)NPAIR * HWN * HWN * 2);
  __bf16* Gs0t  = (__bf16*)carve((size_t)NPAIR * HWN * HWN * 2);
  __bf16* Gs1n  = (__bf16*)carve((size_t)NPAIR * HWN * HWN * 2);
  __bf16* Gs1t  = (__bf16*)carve((size_t)NPAIR * HWN * HWN * 2);
  float*  gj0   = (float*)carve((size_t)NPAIR * 128 * HWN * 4);
  float*  gj1   = (float*)carve((size_t)NPAIR * 128 * HWN * 4);
  float*  ta    = (float*)carve((size_t)NPAIR * 128 * HWN * 4);
  float*  tb    = (float*)carve((size_t)NPAIR * 128 * HWN * 4);
  float*  pa    = (float*)carve((size_t)NPAIR * HWN * 4);
  float*  pb    = (float*)carve((size_t)NPAIR * HWN * 4);
  (void)ws_size; (void)n_in; (void)in_sizes; (void)out_size;

  // One-time fp32 -> bf16 conversions (weights + embed)
  k_cvt<<<(C8N * CC / 4 + 255) / 256, 256, 0, stream>>>(gamma_w, gwb, C8N * CC / 4);
  k_cvt<<<(C8N * CC / 4 + 255) / 256, 256, 0, stream>>>(beta_w,  bwb, C8N * CC / 4);
  k_cvt<<<(128 * 256 / 4 + 255) / 256, 256, 0, stream>>>(gg_w,  ggwb,  128 * 256 / 4);
  k_cvt<<<(128 * 256 / 4 + 255) / 256, 256, 0, stream>>>(tte_w, ttewb, 128 * 256 / 4);
  k_cvt<<<(NIMG * 128 * HWN / 4 + 255) / 256, 256, 0, stream>>>(embed, embedb, NIMG * 128 * HWN / 4);

  // Scalar/vector side chain (cheap)
  k_cp  <<<128, 256, 0, stream>>>(theta_w, theta_b, vect, c0, c1);
  k_mlp1<<<64,  256, 0, stream>>>(cp_w1, cp_b1, c0, c1, h00, h01);
  k_mlp2<<<512, 256, 0, stream>>>(cp_w2, cp_b2, h00, h01, p00, p01);

  // WMMA GEMM chain
  k_gamma_beta<<<dim3(NIMG, C8N / 32), 256, 0, stream>>>(
      re_fm, gwb, bwb, gamma_s, gamma_b, beta_s, beta_b, gammaT, betaN);
  k_gs <<<dim3(NPAIR, HWN / 16), 256, 0, stream>>>(gammaT, betaN, Gs0n, Gs0t, Gs1n, Gs1t);
  k_gg <<<dim3(NPAIR, 128 / 16), 256, 0, stream>>>(ggwb, gg_s, gg_b,
      Gs0n, Gs0t, Gs1n, Gs1t, gj0, gj1);
  k_tte<<<dim3(NPAIR, 128 / 16), 256, 0, stream>>>(ttewb, tte_s, tte_b, embedb, ta, tb);

  // Gates + final elementwise
  k_gate<<<(NPAIR * HWN + 255) / 256, 256, 0, stream>>>(te_w, te_s, te_b, ta, tb, gj0, gj1, pa, pb);
  k_out <<<(NPAIR * CC * HWN / 4 + 255) / 256, 256, 0, stream>>>(featmap, p00, p01, pa, pb, out);
}
